// MatchingNetwork_4690104287237
// MI455X (gfx1250) — matrix operand order repaired and checked by reading, not compiled, not run on hardware
//
#include <hip/hip_runtime.h>

typedef __attribute__((ext_vector_type(16))) _Float16 v16h;
typedef __attribute__((ext_vector_type(8)))  _Float16 v8h;
typedef __attribute__((ext_vector_type(8)))  float    v8f;

#define EPSV 1e-5f

// ---------------------------------------------------------------------------
// f32 -> f16 conversion
// ---------------------------------------------------------------------------
__global__ void cvt_f16(const float* __restrict__ src, _Float16* __restrict__ dst, int n) {
    int i = blockIdx.x * blockDim.x + threadIdx.x;
    if (i < n) dst[i] = (_Float16)src[i];
}

// ---------------------------------------------------------------------------
// Weight swizzle for layer 1 (im2col GEMM, K = CIN*9 padded to KI*32).
// dest d = (((octile*KI + ki)*32 + lane)*16 + half)
// A layout (16x32 f16, contiguous k, same order as B): lane<16: K=half (0..15);
// lane>=16: K=half+16.  M = lane%16.
// ---------------------------------------------------------------------------
template<int CIN, int KI>
__global__ void swizzle_w(const float* __restrict__ w, _Float16* __restrict__ dst, int total) {
    int d = blockIdx.x * blockDim.x + threadIdx.x;
    if (d >= total) return;
    int half   = d & 15;
    int lane   = (d >> 4) & 31;
    int rest   = d >> 9;
    int ki     = rest % KI;
    int octile = rest / KI;
    int oc = octile * 16 + (lane & 15);
    int k  = ki * 32 + ((lane >> 4) & 1) * 16 + half;
    float v = 0.f;
    if (k < CIN * 9) {
        int ci = k / 9, kk = k % 9;
        v = w[((size_t)oc * CIN + ci) * 9 + kk];
    }
    dst[d] = (_Float16)v;
}

// ---------------------------------------------------------------------------
// Weight swizzle for 64->64 layers (shifted-GEMM: K = ci, one GEMM per tap).
// dest d = ((((octile*3+ky)*3+kx)*2 + kb)*32 + lane)*16 + half
// A fragment = W[oc_tile, ci_block(ky,kx)]; K_local per A layout above.
// ---------------------------------------------------------------------------
__global__ void swizzle_w9(const float* __restrict__ w, _Float16* __restrict__ dst, int total) {
    int d = blockIdx.x * blockDim.x + threadIdx.x;
    if (d >= total) return;
    int half   = d & 15;
    int lane   = (d >> 4) & 31;
    int rest   = d >> 9;          // octile*18 + ky*6 + kx*2 + kb
    int kb     = rest & 1;
    int kx     = (rest >> 1) % 3;
    int ky     = (rest / 6) % 3;
    int octile = rest / 18;
    int oc = octile * 16 + (lane & 15);
    int kl = ((lane >> 4) & 1) * 16 + half;  // 0..31
    int ci = kb * 32 + kl;
    dst[d] = (_Float16)w[(((size_t)oc * 64 + ci) * 3 + ky) * 3 + kx];
}

// ---------------------------------------------------------------------------
// Layer-1 conv (3->64) via im2col GEMM: one 16x16x32 WMMA per wave per tile.
// Block = 128 threads (4 waves); wave w -> out-channels [16w,16w+16).
// ---------------------------------------------------------------------------
template<int CIN, int H, int W, int KI>
__global__ __launch_bounds__(128)
void conv_wmma(const _Float16* __restrict__ act, const _Float16* __restrict__ wswz,
               const float* __restrict__ bias, _Float16* __restrict__ out,
               float* __restrict__ stats) {
    constexpr int KTOT = KI * 32;
    constexpr int KUSE = CIN * 9;
    constexpr int HW   = H * W;
    __shared__ __align__(32) _Float16 Bs[16][KTOT + 16];
    __shared__ float ssum[64], ssq[64];

    const int tid  = threadIdx.x;
    const int lane = tid & 31;
    const int wave = tid >> 5;
    const int n    = blockIdx.y;
    const int tile = blockIdx.x;

    if (tid < 64) { ssum[tid] = 0.f; ssq[tid] = 0.f; }

    for (int e = tid; e < 16 * KTOT; e += 128) {
        int px = e / KTOT;
        int k  = e % KTOT;
        float v = 0.f;
        int pix = tile * 16 + px;
        if (k < KUSE && pix < HW) {
            int ci = k / 9, kk = k % 9;
            int ky = kk / 3, kx = kk % 3;
            int y = pix / W, x = pix % W;
            int yi = y + ky - 1, xi = x + kx - 1;
            if (yi >= 0 && yi < H && xi >= 0 && xi < W)
                v = (float)act[(((size_t)n * CIN + ci) * H + yi) * W + xi];
        }
        Bs[px][k] = (_Float16)v;
    }
    __syncthreads();

    const int px = lane & 15;
    const int hi = (lane >> 4) & 1;
    v8f c = {};
    #pragma unroll
    for (int ki = 0; ki < KI; ++ki) {
        v16h a = *(const v16h*)(wswz + ((size_t)(wave * KI + ki) * 32 + lane) * 16);
        v16h b = *(const v16h*)&Bs[px][ki * 32 + hi * 16];
        c = __builtin_amdgcn_wmma_f32_16x16x32_f16(false, a, false, b,
                                                   (short)0, c, false, false);
    }

    int pix = tile * 16 + px;
    bool valid = pix < HW;
    #pragma unroll
    for (int r = 0; r < 8; ++r) {
        int oc = wave * 16 + hi * 8 + r;
        float v = c[r] + bias[oc];
        v = v > 0.f ? v : 0.f;
        if (valid) {
            out[(((size_t)n * 64 + oc) * HW) + pix] = (_Float16)v;
            atomicAdd(&ssum[oc], v);
            atomicAdd(&ssq[oc],  v * v);
        }
    }
    __syncthreads();
    if (tid < 64) {
        atomicAdd(&stats[tid],      ssum[tid]);
        atomicAdd(&stats[64 + tid], ssq[tid]);
    }
}

// ---------------------------------------------------------------------------
// 64->64 conv as 9 shifted GEMMs over K=ci. One block = one 16-pixel strip of
// one output row. LDS holds the raw 3x18x64 activation halo, TRANSPOSED to
// [row][x][ci] so each B fragment is one contiguous 32B LDS read per lane.
// 18 WMMAs per wave (3 ky x 3 kx x 2 ci-blocks).
// ---------------------------------------------------------------------------
template<int H, int W>
__global__ __launch_bounds__(128)
void conv64_wmma(const _Float16* __restrict__ act, const _Float16* __restrict__ wswz,
                 const float* __restrict__ bias, _Float16* __restrict__ out,
                 float* __restrict__ stats) {
    constexpr int SX = (W + 15) / 16;     // strips per row
    __shared__ __align__(32) _Float16 Bs[3][18][64];   // [ky-row][x][ci]
    __shared__ float ssum[64], ssq[64];

    const int tid  = threadIdx.x;
    const int lane = tid & 31;
    const int wave = tid >> 5;
    const int n    = blockIdx.y;
    const int y    = blockIdx.x / SX;
    const int x0   = (blockIdx.x % SX) * 16;

    if (tid < 64) { ssum[tid] = 0.f; ssq[tid] = 0.f; }

    // Stage activation halo: rows y-1..y+1, x0-1..x0+16, all 64 channels.
    // e -> j = e%18 (x), r = (e/18)%3 (row), ci = e/54 : consecutive threads
    // read consecutive x (coalesced within each channel row segment).
    for (int e = tid; e < 3 * 18 * 64; e += 128) {
        int j  = e % 18;
        int r  = (e / 18) % 3;
        int ci = e / 54;
        int yi = y + r - 1;
        int xi = x0 - 1 + j;
        float v = 0.f;
        if (yi >= 0 && yi < H && xi >= 0 && xi < W)
            v = (float)act[(((size_t)n * 64 + ci) * H + yi) * W + xi];
        Bs[r][j][ci] = (_Float16)v;
    }
    __syncthreads();

    const int px = lane & 15;
    const int hi = (lane >> 4) & 1;
    v8f c = {};
    #pragma unroll
    for (int ky = 0; ky < 3; ++ky)
        #pragma unroll
        for (int kx = 0; kx < 3; ++kx)
            #pragma unroll
            for (int kb = 0; kb < 2; ++kb) {
                v16h a = *(const v16h*)(wswz +
                    ((size_t)(((wave * 3 + ky) * 3 + kx) * 2 + kb) * 32 + lane) * 16);
                // B: K = ci_local; lanes 0-15 hold ci kb*32+0..15, lanes 16-31 +16
                v16h b = *(const v16h*)&Bs[ky][px + kx][kb * 32 + hi * 16];
                c = __builtin_amdgcn_wmma_f32_16x16x32_f16(false, a, false, b,
                                                           (short)0, c, false, false);
            }

    int x = x0 + px;
    bool valid = x < W;
    #pragma unroll
    for (int r = 0; r < 8; ++r) {
        int oc = wave * 16 + hi * 8 + r;
        float v = c[r] + bias[oc];
        v = v > 0.f ? v : 0.f;
        if (valid) {
            out[(((size_t)n * 64 + oc) * H + y) * W + x] = (_Float16)v;
            atomicAdd(&ssum[oc], v);
            atomicAdd(&ssq[oc],  v * v);
        }
    }
    __syncthreads();
    if (tid < 64) {
        atomicAdd(&stats[tid],      ssum[tid]);
        atomicAdd(&stats[64 + tid], ssq[tid]);
    }
}

// ---------------------------------------------------------------------------
// BN finalize: stats -> affine (a, c) so BN(x) = a*x + c
// ---------------------------------------------------------------------------
__global__ void bn_finalize(const float* __restrict__ stats, const float* __restrict__ g,
                            const float* __restrict__ be, float cnt, float* __restrict__ coef) {
    int t = threadIdx.x;
    if (t < 64) {
        float mean = stats[t] / cnt;
        float var  = stats[64 + t] / cnt - mean * mean;
        float a    = g[t] * rsqrtf(var + EPSV);
        coef[t]      = a;
        coef[64 + t] = be[t] - mean * a;
    }
}

// ---------------------------------------------------------------------------
// Fused BN-affine + 2x2 maxpool. Writes f16 activation for next layer;
// for the final layer also f32 features laid out as reshape(-1, 1600).
// ---------------------------------------------------------------------------
template<int H, int W, bool LAST>
__global__ void bn_pool(const _Float16* __restrict__ conv, const float* __restrict__ coef,
                        _Float16* __restrict__ outA, float* __restrict__ feat, int total) {
    constexpr int H2 = H / 2, W2 = W / 2;
    int idx = blockIdx.x * blockDim.x + threadIdx.x;
    if (idx >= total) return;
    int x2 = idx % W2;
    int t  = idx / W2;
    int y2 = t % H2; t /= H2;
    int oc = t % 64;
    int n  = t / 64;
    float a = coef[oc], cc = coef[64 + oc];
    float m = -1e30f;
    #pragma unroll
    for (int dy = 0; dy < 2; ++dy)
        #pragma unroll
        for (int dx = 0; dx < 2; ++dx) {
            float v = a * (float)conv[(((size_t)n * 64 + oc) * H + 2 * y2 + dy) * W + 2 * x2 + dx] + cc;
            m = v > m ? v : m;
        }
    outA[(((size_t)n * 64 + oc) * H2 + y2) * W2 + x2] = (_Float16)m;
    if (LAST)
        feat[(size_t)n * 1600 + oc * 25 + y2 * 5 + x2] = m;
}

// ---------------------------------------------------------------------------
// Simplex similarity: one block per (b, way). 128 threads.
// ---------------------------------------------------------------------------
__global__ __launch_bounds__(128)
void simplex(const float* __restrict__ supF, const float* __restrict__ smpF,
             float* __restrict__ out) {
    __shared__ float red[15][129];
    __shared__ float Gs[15];
    __shared__ float sVolA;
    const int tid = threadIdx.x;
    const int b   = blockIdx.x / 5;
    const int way = blockIdx.x % 5;
    const float* s0 = supF + ((size_t)(b * 25 + way * 5)) * 1600;

    float p[10];
    #pragma unroll
    for (int e = 0; e < 10; ++e) p[e] = 0.f;
    for (int d = tid; d < 1600; d += 128) {
        float v0 = s0[d];
        float a0 = s0[1600 + d] - v0, a1 = s0[3200 + d] - v0;
        float a2 = s0[4800 + d] - v0, a3 = s0[6400 + d] - v0;
        p[0] += a0 * a0; p[1] += a0 * a1; p[2] += a0 * a2; p[3] += a0 * a3;
        p[4] += a1 * a1; p[5] += a1 * a2; p[6] += a1 * a3;
        p[7] += a2 * a2; p[8] += a2 * a3; p[9] += a3 * a3;
    }
    #pragma unroll
    for (int e = 0; e < 10; ++e) red[e][tid] = p[e];
    __syncthreads();
    if (tid < 10) {
        float s = 0.f;
        for (int j = 0; j < 128; ++j) s += red[tid][j];
        Gs[tid] = s;
    }
    __syncthreads();
    if (tid == 0) {
        float M[4][4] = {{Gs[0], Gs[1], Gs[2], Gs[3]},
                         {Gs[1], Gs[4], Gs[5], Gs[6]},
                         {Gs[2], Gs[5], Gs[7], Gs[8]},
                         {Gs[3], Gs[6], Gs[8], Gs[9]}};
        float det = 1.f;
        for (int i = 0; i < 4; ++i) {
            float piv = M[i][i]; det *= piv;
            float inv = 1.f / piv;
            for (int r = i + 1; r < 4; ++r) {
                float f = M[r][i] * inv;
                for (int c2 = i; c2 < 4; ++c2) M[r][c2] -= f * M[i][c2];
            }
        }
        sVolA = det;
    }

    for (int q = 0; q < 15; ++q) {
        const float* mq = smpF + ((size_t)(b * 15 + q)) * 1600;
        float t[15];
        #pragma unroll
        for (int e = 0; e < 15; ++e) t[e] = 0.f;
        for (int d = tid; d < 1600; d += 128) {
            float m  = mq[d];
            float r0 = s0[d] - m,        r1 = s0[1600 + d] - m;
            float r2 = s0[3200 + d] - m, r3 = s0[4800 + d] - m;
            float r4 = s0[6400 + d] - m;
            t[0]  += r0 * r0; t[1]  += r0 * r1; t[2]  += r0 * r2; t[3]  += r0 * r3; t[4]  += r0 * r4;
            t[5]  += r1 * r1; t[6]  += r1 * r2; t[7]  += r1 * r3; t[8]  += r1 * r4;
            t[9]  += r2 * r2; t[10] += r2 * r3; t[11] += r2 * r4;
            t[12] += r3 * r3; t[13] += r3 * r4;
            t[14] += r4 * r4;
        }
        __syncthreads();
        #pragma unroll
        for (int e = 0; e < 15; ++e) red[e][tid] = t[e];
        __syncthreads();
        if (tid < 15) {
            float s = 0.f;
            for (int j = 0; j < 128; ++j) s += red[tid][j];
            Gs[tid] = s;
        }
        __syncthreads();
        if (tid == 0) {
            float M[5][5];
            M[0][0] = Gs[0];  M[0][1] = Gs[1];  M[0][2] = Gs[2];  M[0][3] = Gs[3];  M[0][4] = Gs[4];
            M[1][1] = Gs[5];  M[1][2] = Gs[6];  M[1][3] = Gs[7];  M[1][4] = Gs[8];
            M[2][2] = Gs[9];  M[2][3] = Gs[10]; M[2][4] = Gs[11];
            M[3][3] = Gs[12]; M[3][4] = Gs[13];
            M[4][4] = Gs[14];
            for (int i = 1; i < 5; ++i)
                for (int j = 0; j < i; ++j) M[i][j] = M[j][i];
            float det = 1.f;
            for (int i = 0; i < 5; ++i) {
                float piv = M[i][i]; det *= piv;
                float inv = 1.f / piv;
                for (int r = i + 1; r < 5; ++r) {
                    float f = M[r][i] * inv;
                    for (int c2 = i; c2 < 5; ++c2) M[r][c2] -= f * M[i][c2];
                }
            }
            out[((size_t)(b * 15 + q)) * 5 + way] = -(det / sVolA);
        }
    }
}

// ---------------------------------------------------------------------------
// Host launcher
// ---------------------------------------------------------------------------
extern "C" void kernel_launch(void* const* d_in, const int* in_sizes, int n_in,
                              void* d_out, int out_size, void* d_ws, size_t ws_size,
                              hipStream_t stream) {
    (void)in_sizes; (void)n_in; (void)out_size; (void)ws_size;

    const float* support = (const float*)d_in[0];
    // d_in[1] = support_label : unused by reference
    const float* sample  = (const float*)d_in[2];
    const float* Wc[4] = {(const float*)d_in[3],  (const float*)d_in[7],
                          (const float*)d_in[11], (const float*)d_in[15]};
    const float* Bc[4] = {(const float*)d_in[4],  (const float*)d_in[8],
                          (const float*)d_in[12], (const float*)d_in[16]};
    const float* Gc[4] = {(const float*)d_in[5],  (const float*)d_in[9],
                          (const float*)d_in[13], (const float*)d_in[17]};
    const float* Ec[4] = {(const float*)d_in[6],  (const float*)d_in[10],
                          (const float*)d_in[14], (const float*)d_in[18]};
    float* outP = (float*)d_out;

    char* ws = (char*)d_ws;
    size_t off = 0;
    auto carve = [&](size_t bytes) -> char* {
        char* p = ws + off;
        off = (off + bytes + 255) & ~(size_t)255;
        return p;
    };
    _Float16* wz1  = (_Float16*)carve(2048 * sizeof(_Float16));
    _Float16* wz2  = (_Float16*)carve(36864 * sizeof(_Float16));
    _Float16* wz3  = (_Float16*)carve(36864 * sizeof(_Float16));
    _Float16* wz4  = (_Float16*)carve(36864 * sizeof(_Float16));
    float*    stats = (float*)carve(128 * sizeof(float));
    float*    coef  = (float*)carve(128 * sizeof(float));
    float*    supF  = (float*)carve((size_t)8 * 25 * 1600 * sizeof(float));
    float*    smpF  = (float*)carve((size_t)8 * 15 * 1600 * sizeof(float));
    _Float16* bufA  = (_Float16*)carve((size_t)200 * 64 * 42 * 42 * sizeof(_Float16));
    _Float16* bufC  = (_Float16*)carve((size_t)200 * 64 * 84 * 84 * sizeof(_Float16));

    swizzle_w<3, 1><<<(2048 + 255) / 256, 256, 0, stream>>>(Wc[0], wz1, 2048);
    swizzle_w9<<<(36864 + 255) / 256, 256, 0, stream>>>(Wc[1], wz2, 36864);
    swizzle_w9<<<(36864 + 255) / 256, 256, 0, stream>>>(Wc[2], wz3, 36864);
    swizzle_w9<<<(36864 + 255) / 256, 256, 0, stream>>>(Wc[3], wz4, 36864);

    auto pass = [&](const float* img, int N, float* feat) {
        int nelt = N * 3 * 84 * 84;
        cvt_f16<<<(nelt + 255) / 256, 256, 0, stream>>>(img, bufA, nelt);

        // ---- layer 1: 3 -> 64, 84x84 (im2col GEMM, 1 WMMA/wave/tile) ----
        hipMemsetAsync(stats, 0, 128 * sizeof(float), stream);
        conv_wmma<3, 84, 84, 1><<<dim3(441, N), 128, 0, stream>>>(bufA, wz1, Bc[0], bufC, stats);
        bn_finalize<<<1, 64, 0, stream>>>(stats, Gc[0], Ec[0], (float)N * 7056.f, coef);
        int t1 = N * 64 * 42 * 42;
        bn_pool<84, 84, false><<<(t1 + 255) / 256, 256, 0, stream>>>(bufC, coef, bufA, nullptr, t1);

        // ---- layer 2: 64 -> 64, 42x42 (shifted GEMM) ----
        hipMemsetAsync(stats, 0, 128 * sizeof(float), stream);
        conv64_wmma<42, 42><<<dim3(42 * 3, N), 128, 0, stream>>>(bufA, wz2, Bc[1], bufC, stats);
        bn_finalize<<<1, 64, 0, stream>>>(stats, Gc[1], Ec[1], (float)N * 1764.f, coef);
        int t2 = N * 64 * 21 * 21;
        bn_pool<42, 42, false><<<(t2 + 255) / 256, 256, 0, stream>>>(bufC, coef, bufA, nullptr, t2);

        // ---- layer 3: 64 -> 64, 21x21 ----
        hipMemsetAsync(stats, 0, 128 * sizeof(float), stream);
        conv64_wmma<21, 21><<<dim3(21 * 2, N), 128, 0, stream>>>(bufA, wz3, Bc[2], bufC, stats);
        bn_finalize<<<1, 64, 0, stream>>>(stats, Gc[2], Ec[2], (float)N * 441.f, coef);
        int t3 = N * 64 * 10 * 10;
        bn_pool<21, 21, false><<<(t3 + 255) / 256, 256, 0, stream>>>(bufC, coef, bufA, nullptr, t3);

        // ---- layer 4: 64 -> 64, 10x10 ----
        hipMemsetAsync(stats, 0, 128 * sizeof(float), stream);
        conv64_wmma<10, 10><<<dim3(10 * 1, N), 128, 0, stream>>>(bufA, wz4, Bc[3], bufC, stats);
        bn_finalize<<<1, 64, 0, stream>>>(stats, Gc[3], Ec[3], (float)N * 100.f, coef);
        int t4 = N * 64 * 5 * 5;
        bn_pool<10, 10, true><<<(t4 + 255) / 256, 256, 0, stream>>>(bufC, coef, bufA, feat, t4);
    };

    pass(support, 200, supF);   // B * WAY * SHOT = 200 images
    pass(sample, 120, smpF);    // B * QUIRY     = 120 images

    simplex<<<40, 128, 0, stream>>>(supF, smpF, outP);
}
